// MoE_61993557950953
// MI455X (gfx1250) — matrix operand order repaired and checked by reading, not compiled, not run on hardware
//
#include <hip/hip_runtime.h>
#include <hip/hip_bf16.h>
#include <math.h>

typedef __attribute__((ext_vector_type(16))) __bf16 v16bf;
typedef __attribute__((ext_vector_type(8)))  float  v8f;

#define B_TOK 16384
#define DIM   512
#define NEXP  16
#define HDIM  128
#define ODIM  512
#define LN_EPS 1e-5f

union ABfrag { v16bf v; uint4 q[2]; };

static __device__ __forceinline__ v8f wmma_bf16(v16bf a, v16bf b, v8f c) {
  return __builtin_amdgcn_wmma_f32_16x16x32_bf16(false, a, false, b, (short)0, c, false, false);
}

// ---------------- conversion / transpose kernels ----------------

__global__ void cvt_x_kernel(const float* __restrict__ x, __bf16* __restrict__ xb) {
  int i = (blockIdx.x * 256 + threadIdx.x) * 4;
  float4 v = *(const float4*)(x + i);
  __bf16 o[4] = { (__bf16)v.x, (__bf16)v.y, (__bf16)v.z, (__bf16)v.w };
  *(uint2*)(xb + i) = *(const uint2*)o;
}

// W1 [E, D, H] (row-major) -> w1t [E, H, D] bf16
__global__ void tr_w1_kernel(const float* __restrict__ W1, __bf16* __restrict__ w1t) {
  int i = blockIdx.x * 256 + threadIdx.x;      // over E*D*H = 1048576
  int e = i >> 16;                             // D*H = 65536
  int r = i & 65535;
  int d = r >> 7;                              // / H
  int h = r & 127;
  w1t[(e << 16) + h * DIM + d] = (__bf16)W1[i];
}

// W2 [E, H, O] -> w2t [E, O, H] bf16
__global__ void tr_w2_kernel(const float* __restrict__ W2, __bf16* __restrict__ w2t) {
  int i = blockIdx.x * 256 + threadIdx.x;      // over E*H*O = 1048576
  int e = i >> 16;                             // H*O = 65536
  int r = i & 65535;
  int h = r >> 9;                              // / O
  int o = r & 511;
  w2t[(e << 16) + o * HDIM + h] = (__bf16)W2[i];
}

// ---------------- gating: scores, top-2, routing lists, entropy ----------------

__global__ void __launch_bounds__(256) gate_kernel(
    const float* __restrict__ x, const float* __restrict__ gW, const float* __restrict__ gb,
    int* __restrict__ counts, int* __restrict__ list, float* __restrict__ gatew,
    float* __restrict__ entAcc)
{
  __shared__ float sgw[DIM * NEXP];            // 32 KB
  const int tid = threadIdx.x;
  for (int i = tid; i < DIM * NEXP; i += 256) sgw[i] = gW[i];
  __syncthreads();

  const int wave = tid >> 5, lane = tid & 31;
  const int t = blockIdx.x * 8 + wave;

  float p[NEXP];
  #pragma unroll
  for (int e = 0; e < NEXP; ++e) p[e] = 0.f;

  const float* xrow = x + (size_t)t * DIM;
  for (int d = lane; d < DIM; d += 32) {
    const float xv = xrow[d];
    const float4* w4 = (const float4*)(sgw + d * NEXP);
    #pragma unroll
    for (int j = 0; j < 4; ++j) {
      float4 w = w4[j];
      p[4*j+0] += xv * w.x; p[4*j+1] += xv * w.y;
      p[4*j+2] += xv * w.z; p[4*j+3] += xv * w.w;
    }
  }
  #pragma unroll
  for (int e = 0; e < NEXP; ++e) {
    #pragma unroll
    for (int off = 16; off; off >>= 1) p[e] += __shfl_xor(p[e], off, 32);
    p[e] += gb[e];
  }

  if (lane == 0) {
    float m1 = -1e30f; int i1 = 0;
    for (int e = 0; e < NEXP; ++e) if (p[e] > m1) { m1 = p[e]; i1 = e; }
    float m2 = -1e30f; int i2 = 0;
    for (int e = 0; e < NEXP; ++e) if (e != i1 && p[e] > m2) { m2 = p[e]; i2 = e; }
    const float w1 = 1.f / (1.f + __expf(m2 - m1));   // softmax over top-2
    const float w2 = 1.f - w1;
    int p1 = atomicAdd(&counts[i1], 1);
    list[i1 * B_TOK + p1] = t * 2 + 0;  gatew[t * 2 + 0] = w1;
    int p2 = atomicAdd(&counts[i2], 1);
    list[i2 * B_TOK + p2] = t * 2 + 1;  gatew[t * 2 + 1] = w2;
    // full-16 softmax entropy
    float Z = 0.f;
    for (int e = 0; e < NEXP; ++e) Z += __expf(p[e] - m1);
    const float lZ = __logf(Z);
    float ent = 0.f;
    for (int e = 0; e < NEXP; ++e) {
      float lp = p[e] - m1 - lZ;
      ent -= __expf(lp) * lp;
    }
    atomicAdd(entAcc, ent);
  }
}

// ---------------- routed expert compute: GEMM1 -> GELU -> LN -> GEMM2 -> scatter ----------------

__global__ void __launch_bounds__(256) expert_kernel(
    const __bf16* __restrict__ xb, const __bf16* __restrict__ w1t, const __bf16* __restrict__ w2t,
    const float* __restrict__ b1, const float* __restrict__ lng, const float* __restrict__ lnb,
    const float* __restrict__ b2, const int* __restrict__ counts, const int* __restrict__ list,
    const float* __restrict__ gatew, float* __restrict__ out)
{
  const int e    = blockIdx.y;
  const int cnt  = counts[e];
  const int tile = blockIdx.x;
  if (tile * 16 >= cnt) return;

  __shared__ int   rowTok[16];
  __shared__ float rowG[16];
  __shared__ __align__(16) float  hS[16 * HDIM];    // 8 KB fp32 h
  __shared__ __align__(16) __bf16 hB[16 * HDIM];    // 4 KB bf16 normalized h

  const int tid = threadIdx.x;
  if (tid < 16) {
    int idx = tile * 16 + tid;
    if (idx < cnt) { int pk = list[e * B_TOK + idx]; rowTok[tid] = pk >> 1; rowG[tid] = gatew[pk]; }
    else           { rowTok[tid] = 0;                rowG[tid] = 0.f; }
  }
  __syncthreads();

  const int wave = tid >> 5, lane = tid & 31;
  const int half = lane >> 4, l16 = lane & 15;

  // ---- GEMM1: [16 x 512] @ [512 x 128]; wave owns column tile `wave`
  const __bf16* aBase = xb + (size_t)rowTok[l16] * DIM;
  const __bf16* bBase = w1t + ((size_t)e * HDIM + wave * 16 + l16) * DIM;
  v8f acc = {};
  #pragma unroll
  for (int k = 0; k < DIM / 32; ++k) {
    const int kb = k * 32;
    ABfrag a, b;
    a.q[0] = *(const uint4*)(aBase + kb + half * 16);       // K = kb+half*16 .. +7
    a.q[1] = *(const uint4*)(aBase + kb + half * 16 + 8);   // K = kb+half*16+8 .. +15
    b.q[0] = *(const uint4*)(bBase + kb + half * 16);       // K = kb+half*16 .. +15 (contiguous)
    b.q[1] = *(const uint4*)(bBase + kb + half * 16 + 8);
    acc = wmma_bf16(a.v, b.v, acc);
  }
  // epilogue1: bias + exact GELU -> LDS fp32
  {
    const int col = wave * 16 + l16;
    const float bias = b1[e * HDIM + col];
    #pragma unroll
    for (int r = 0; r < 8; ++r) {
      const int row = r + half * 8;
      float v = acc[r] + bias;
      v = 0.5f * v * (1.f + erff(v * 0.70710678118654752f));
      hS[row * HDIM + col] = v;
    }
  }
  __syncthreads();

  // ---- LayerNorm over H=128 per row; 8 waves x 2 rows
  #pragma unroll
  for (int rr = 0; rr < 2; ++rr) {
    const int row = wave * 2 + rr;
    float vals[4], s = 0.f, sq = 0.f;
    #pragma unroll
    for (int j = 0; j < 4; ++j) {
      float v = hS[row * HDIM + lane + 32 * j];
      vals[j] = v; s += v; sq += v * v;
    }
    #pragma unroll
    for (int off = 16; off; off >>= 1) { s += __shfl_xor(s, off, 32); sq += __shfl_xor(sq, off, 32); }
    const float mu  = s * (1.f / HDIM);
    const float inv = rsqrtf(sq * (1.f / HDIM) - mu * mu + LN_EPS);
    #pragma unroll
    for (int j = 0; j < 4; ++j) {
      const int col = lane + 32 * j;
      float v = (vals[j] - mu) * inv * lng[e * HDIM + col] + lnb[e * HDIM + col];
      hB[row * HDIM + col] = (__bf16)v;
    }
  }
  __syncthreads();

  // ---- GEMM2: [16 x 128] @ [128 x 512]; wave owns n-tiles {wave, wave+8, wave+16, wave+24}
  v8f acc2[4] = {};
  #pragma unroll
  for (int k = 0; k < HDIM / 32; ++k) {
    const int kb = k * 32;
    ABfrag a;
    a.q[0] = *(const uint4*)(hB + l16 * HDIM + kb + half * 16);
    a.q[1] = *(const uint4*)(hB + l16 * HDIM + kb + half * 16 + 8);
    #pragma unroll
    for (int j = 0; j < 4; ++j) {
      const __bf16* bb = w2t + ((size_t)e * ODIM + (wave + 8 * j) * 16 + l16) * HDIM + kb + half * 16;
      ABfrag b;
      b.q[0] = *(const uint4*)bb;
      b.q[1] = *(const uint4*)(bb + 8);
      acc2[j] = wmma_bf16(a.v, b.v, acc2[j]);
    }
  }
  // epilogue2: bias, gate weight, scatter-add (each token has exactly 2 expert contributions)
  #pragma unroll
  for (int j = 0; j < 4; ++j) {
    const int col = (wave + 8 * j) * 16 + l16;
    const float bias = b2[e * ODIM + col];
    #pragma unroll
    for (int r = 0; r < 8; ++r) {
      const int row = r + half * 8;
      const float v = rowG[row] * (acc2[j][r] + bias);
      atomicAdd(&out[(size_t)rowTok[row] * ODIM + col], v);
    }
  }
}

// ---------------- aux loss finalize ----------------

__global__ void finalize_kernel(const int* __restrict__ counts, const float* __restrict__ entAcc,
                                float* __restrict__ auxOut) {
  if (threadIdx.x == 0) {
    float lb = 0.f;
    for (int e = 0; e < NEXP; ++e) {
      float u = (float)counts[e] * (1.f / B_TOK);
      float d = u - 1.f / NEXP;
      lb += d * d;
    }
    lb *= (1.f / NEXP);
    auxOut[0] = lb - 0.1f * (entAcc[0] * (1.f / B_TOK));
  }
}

// ---------------- launcher ----------------

extern "C" void kernel_launch(void* const* d_in, const int* in_sizes, int n_in,
                              void* d_out, int out_size, void* d_ws, size_t ws_size,
                              hipStream_t stream) {
  const float* x   = (const float*)d_in[0];
  const float* gW  = (const float*)d_in[1];
  const float* gb  = (const float*)d_in[2];
  const float* W1  = (const float*)d_in[3];
  const float* b1  = (const float*)d_in[4];
  const float* lng = (const float*)d_in[5];
  const float* lnb = (const float*)d_in[6];
  const float* W2  = (const float*)d_in[7];
  const float* b2  = (const float*)d_in[8];
  float* out = (float*)d_out;

  char* ws = (char*)d_ws;
  __bf16* xb     = (__bf16*)ws;                                   // 16 MB
  __bf16* w1t    = (__bf16*)(ws + (size_t)16 * 1048576);          //  2 MB
  __bf16* w2t    = (__bf16*)(ws + (size_t)18 * 1048576);          //  2 MB
  const size_t base = (size_t)20 * 1048576;
  int*    counts = (int*)  (ws + base);                           // 64 B
  float*  entAcc = (float*)(ws + base + 64);                      // 4 B
  int*    list   = (int*)  (ws + base + 128);                     // 1 MB
  float*  gatew  = (float*)(ws + base + 128 + (size_t)NEXP * B_TOK * 4); // 128 KB

  // zero accumulators and output (atomically accumulated)
  hipMemsetAsync(ws + base, 0, 128, stream);
  hipMemsetAsync(d_out, 0, (size_t)out_size * sizeof(float), stream);

  cvt_x_kernel <<< (B_TOK * DIM / 4) / 256, 256, 0, stream >>> (x, xb);
  tr_w1_kernel <<< (NEXP * DIM * HDIM) / 256, 256, 0, stream >>> (W1, w1t);
  tr_w2_kernel <<< (NEXP * HDIM * ODIM) / 256, 256, 0, stream >>> (W2, w2t);

  gate_kernel  <<< B_TOK / 8, 256, 0, stream >>> (x, gW, gb, counts, list, gatew, entAcc);

  expert_kernel<<< dim3(B_TOK / 16, NEXP), 256, 0, stream >>> (
      xb, w1t, w2t, b1, lng, lnb, b2, counts, list, gatew, out);

  finalize_kernel<<< 1, 32, 0, stream >>> (counts, entAcc, out + (size_t)B_TOK * ODIM);
}
